// DecoderMultiHeadedAttention_90082644066830
// MI455X (gfx1250) — matrix operand order repaired and checked by reading, not compiled, not run on hardware
//
#include <hip/hip_runtime.h>

#define B_ 4
#define S_ 1024
#define D_ 1024
#define H_ 16
#define DH_ 64

typedef __attribute__((ext_vector_type(16))) __bf16        v16bf;
typedef __attribute__((ext_vector_type(2)))  __bf16        v2bf;
typedef __attribute__((ext_vector_type(8)))  float         v8f;
typedef __attribute__((ext_vector_type(4)))  unsigned int  u32x4;
typedef __attribute__((ext_vector_type(2)))  unsigned int  u32x2;
typedef __attribute__((ext_vector_type(4)))  float         f32x4;
typedef int v4i_gcc __attribute__((vector_size(16)));      // matches builtin proto

union FragBF {            // one WMMA bf16 operand: 16 bf16 per lane = 32 bytes
    v16bf v;
    u32x4 u[2];
};

// ---- CDNA5 async global->LDS path (guarded; falls back to register bounce)
#if defined(__has_builtin)
#if __has_builtin(__builtin_amdgcn_global_load_async_to_lds_b128) && \
    __has_builtin(__builtin_amdgcn_s_wait_asynccnt)
#define USE_ASYNC 1
#endif
#if __has_builtin(__builtin_amdgcn_cvt_pk_bf16_f32)
#define USE_PK_CVT 1
#endif
#endif
#ifndef USE_ASYNC
#define USE_ASYNC 0
#endif
#ifndef USE_PK_CVT
#define USE_PK_CVT 0
#endif

#if USE_ASYNC
#define ASYNC_WAIT(n) __builtin_amdgcn_s_wait_asynccnt(n)
#else
#define ASYNC_WAIT(n) ((void)0)
#endif

__device__ __forceinline__ void cp16(void* lds, const void* glb) {
#if USE_ASYNC
    __builtin_amdgcn_global_load_async_to_lds_b128(
        (__attribute__((address_space(1))) v4i_gcc*)glb,
        (__attribute__((address_space(3))) v4i_gcc*)lds, 0, 0);
#else
    *(u32x4*)lds = *(const u32x4*)glb;
#endif
}

// pack two f32 into two bf16 (RNE) in one dword
__device__ __forceinline__ unsigned pack2(float a, float b) {
#if USE_PK_CVT
    union { v2bf v; unsigned u; } c;
    c.v = __builtin_amdgcn_cvt_pk_bf16_f32(a, b);
    return c.u;
#else
    unsigned ua = __float_as_uint(a);
    ua += 0x7FFFu + ((ua >> 16) & 1u);
    unsigned ub = __float_as_uint(b);
    ub += 0x7FFFu + ((ub >> 16) & 1u);
    return (ua >> 16) | (ub & 0xFFFF0000u);
#endif
}

__device__ __forceinline__ unsigned short f2bf(float f) {
    return (unsigned short)(pack2(f, f) & 0xFFFFu);
}

__device__ __forceinline__ v8f wmma_bf16(const FragBF& a, const FragBF& b, v8f c) {
    return __builtin_amdgcn_wmma_f32_16x16x32_bf16(
        false, a.v, false, b.v, (short)0, c, false, false);
}

// ---------------------------------------------------------------------------
// Kernel 1a: convert the four weight matrices fp32 -> bf16.  grid=(1024, 4)
// Kernel 1b: convert query/key/value fp32 -> bf16.           grid=(4096, 3)
// One-time passes; hoists ALL activation conversion out of the GEMM K-loops
// (previously each X element was converted once per n-block = 8x redundant).
// ---------------------------------------------------------------------------
__global__ __launch_bounds__(256) void convert_weights(
    const float* __restrict__ wq, const float* __restrict__ wk,
    const float* __restrict__ wv, const float* __restrict__ wo,
    unsigned short* __restrict__ out)
{
    const int m = blockIdx.y;
    const float* src = (m == 0) ? wq : (m == 1) ? wk : (m == 2) ? wv : wo;
    size_t i = ((size_t)blockIdx.x * 256 + threadIdx.x) * 4;
    f32x4 vv = *(const f32x4*)(src + i);
    unsigned short* o = out + (size_t)m * D_ * D_ + i;
    u32x2 pk;
    pk.x = pack2(vv.x, vv.y);
    pk.y = pack2(vv.z, vv.w);
    *(u32x2*)o = pk;
}

__global__ __launch_bounds__(256) void convert_inputs(
    const float* __restrict__ q, const float* __restrict__ k,
    const float* __restrict__ v, unsigned short* __restrict__ out)
{
    const int m = blockIdx.y;
    const float* src = (m == 0) ? q : (m == 1) ? k : v;
    size_t i = ((size_t)blockIdx.x * 256 + threadIdx.x) * 4;
    f32x4 vv = *(const f32x4*)(src + i);
    unsigned short* o = out + (size_t)m * B_ * S_ * D_ + i;
    u32x2 pk;
    pk.x = pack2(vv.x, vv.y);
    pk.y = pack2(vv.z, vv.w);
    *(u32x2*)o = pk;
}

// ---------------------------------------------------------------------------
// Kernel 2: QKV projection.  C[4096x1024] = X @ W^T + bias  (NT GEMM, all
// bf16 operands).  blockIdx.z picks projection (0:Q 1:K 2:V).  Block tile
// 128x128x32, 8 waves in 4(m-pair) x 2(n-half); each wave 2 A-frags x 4
// B-frags = 8 WMMA per K-step.  Async double-buffered LDS staging.
// Q,K stored [b][h][s][dh] bf16; V stored transposed [b][h][dh][s] bf16.
// ---------------------------------------------------------------------------
__global__ __launch_bounds__(256) void qkv_gemm(
    const unsigned short* __restrict__ Xbf3, const unsigned short* __restrict__ Wbf,
    const float* __restrict__ bq, const float* __restrict__ bk,
    const float* __restrict__ bv,
    unsigned short* __restrict__ Qp, unsigned short* __restrict__ Kp,
    unsigned short* __restrict__ Vt)
{
    __shared__ unsigned short ldsX[2][128 * 32];   // 8 KB x2
    __shared__ unsigned short ldsW[2][128 * 32];   // 8 KB x2

    const int proj = blockIdx.z;
    const unsigned short* X = Xbf3 + (size_t)proj * B_ * S_ * D_;
    const unsigned short* W = Wbf + (size_t)proj * D_ * D_;
    const float* bias = (proj == 0) ? bq : (proj == 1) ? bk : bv;

    const int tid  = threadIdx.x;
    const int wave = tid >> 5;
    const int lane = tid & 31;
    const int ln   = lane & 15;
    const int hh   = lane >> 4;
    const int m0   = blockIdx.x * 128;
    const int n0   = blockIdx.y * 128;
    const int mq   = wave & 3;   // m-pair: m-tiles 2*mq, 2*mq+1
    const int nh   = wave >> 2;  // n-half: n-tiles nh*4 .. nh*4+3

    const int xrow = tid >> 2, xo = (tid & 3) * 8;   // + i*64 rows, i=0..1

    v8f acc[2][4];
    for (int a = 0; a < 2; ++a)
        for (int i = 0; i < 4; ++i)
            for (int g = 0; g < 8; ++g) acc[a][i][g] = 0.0f;

    for (int i = 0; i < 2; ++i) {
        cp16(&ldsX[0][(xrow + i * 64) * 32 + xo],
             X + (size_t)(m0 + xrow + i * 64) * D_ + xo);
        cp16(&ldsW[0][(xrow + i * 64) * 32 + xo],
             W + (size_t)(n0 + xrow + i * 64) * D_ + xo);
    }

    for (int it = 0; it < D_ / 32; ++it) {
        const int cur = it & 1;
        if (it + 1 < D_ / 32) {
            const int nk = (it + 1) * 32, nb = cur ^ 1;
            for (int i = 0; i < 2; ++i) {
                cp16(&ldsX[nb][(xrow + i * 64) * 32 + xo],
                     X + (size_t)(m0 + xrow + i * 64) * D_ + nk + xo);
                cp16(&ldsW[nb][(xrow + i * 64) * 32 + xo],
                     W + (size_t)(n0 + xrow + i * 64) * D_ + nk + xo);
            }
            ASYNC_WAIT(4);
        } else {
            ASYNC_WAIT(0);
        }
        __syncthreads();

        const unsigned short* xb = ldsX[cur];
        const unsigned short* wb = ldsW[cur];
        FragBF af[2];
        for (int a = 0; a < 2; ++a) {   // A: lane=M row; K={8h..8h+7,16+8h..}
            const unsigned short* ab = xb + ((mq * 2 + a) * 16 + ln) * 32;
            af[a].u[0] = *(const u32x4*)(ab + 8 * hh);
            af[a].u[1] = *(const u32x4*)(ab + 16 + 8 * hh);
        }
        for (int i = 0; i < 4; ++i) {
            FragBF bf;   // B: lane = N col (W row), K={8h..8h+7,16+8h..}
            const unsigned short* bb = wb + (nh * 64 + i * 16 + ln) * 32;
            bf.u[0] = *(const u32x4*)(bb + 8 * hh);
            bf.u[1] = *(const u32x4*)(bb + 16 + 8 * hh);
            acc[0][i] = wmma_bf16(af[0], bf, acc[0][i]);
            acc[1][i] = wmma_bf16(af[1], bf, acc[1][i]);
        }
        __syncthreads();
    }

    // epilogue: bias add, head-split store (V transposed)
    for (int a = 0; a < 2; ++a)
        for (int i = 0; i < 4; ++i) {
            int n = n0 + nh * 64 + i * 16 + ln;
            float bn = bias[n];
            int hcol = n >> 6, dh = n & 63;
            for (int g = 0; g < 8; ++g) {
                int m = m0 + (mq * 2 + a) * 16 + g + 8 * hh;
                int bb_ = m >> 10, s = m & 1023;
                unsigned short o = f2bf(acc[a][i][g] + bn);
                if (proj == 2)
                    Vt[(((size_t)bb_ * H_ + hcol) * DH_ + dh) * S_ + s] = o;
                else if (proj == 0)
                    Qp[(((size_t)bb_ * H_ + hcol) * S_ + s) * DH_ + dh] = o;
                else
                    Kp[(((size_t)bb_ * H_ + hcol) * S_ + s) * DH_ + dh] = o;
            }
        }
}

// ---------------------------------------------------------------------------
// Kernel 3: flash attention.  grid = (B*H, S/128).  8 waves; each wave owns a
// 16-row q tile.  kv streamed in steps of 32 with double-buffered async K/V
// tiles; online softmax via 16-lane shfl reductions; P relayout C->A through
// a per-wave 1 KB LDS bounce.
// ---------------------------------------------------------------------------
__global__ __launch_bounds__(256) void attn_kernel(
    const unsigned short* __restrict__ Qp, const unsigned short* __restrict__ Kp,
    const unsigned short* __restrict__ Vt, const int* __restrict__ mask,
    unsigned short* __restrict__ attn)
{
    __shared__ unsigned short ldsK[2][32 * 64];     // [kv][dh]  4 KB x2
    __shared__ unsigned short ldsV[2][64 * 32];     // [dh][kv]  4 KB x2
    __shared__ unsigned short ldsP[8][16 * 32];     // per-wave P tiles, 8 KB

    const int tid  = threadIdx.x;
    const int wave = tid >> 5;
    const int lane = tid & 31;
    const int ln   = lane & 15;
    const int hh   = lane >> 4;
    const int bh   = blockIdx.x;
    const int b    = bh >> 4;
    const int h    = bh & 15;
    const int q0   = blockIdx.y * 128 + wave * 16;
    const float scale = 0.125f;  // 1/sqrt(64)

    const int krow = tid >> 3, koff = (tid & 7) * 8;   // K tile chunk
    const int vrow = tid >> 2, voff = (tid & 3) * 8;   // V tile chunk
    const unsigned short* Kb = Kp + (size_t)bh * S_ * DH_;
    const unsigned short* Vb = Vt + (size_t)bh * DH_ * S_;

    // Q fragments (dh windows 0-31, 32-63), loaded once
    FragBF qf[2];
    {
        const unsigned short* qb = Qp + ((size_t)bh * S_ + q0 + ln) * DH_;
        qf[0].u[0] = *(const u32x4*)(qb + 8 * hh);
        qf[0].u[1] = *(const u32x4*)(qb + 16 + 8 * hh);
        qf[1].u[0] = *(const u32x4*)(qb + 32 + 8 * hh);
        qf[1].u[1] = *(const u32x4*)(qb + 48 + 8 * hh);
    }

    v8f O[4];
    for (int d = 0; d < 4; ++d)
        for (int g = 0; g < 8; ++g) O[d][g] = 0.0f;
    float mrow[8], lrow[8];
    for (int g = 0; g < 8; ++g) { mrow[g] = -1e30f; lrow[g] = 0.0f; }

    // prologue: stage kv tile 0 into buffer 0
    cp16(&ldsK[0][krow * 64 + koff], Kb + (size_t)krow * DH_ + koff);
    cp16(&ldsV[0][vrow * 32 + voff], Vb + (size_t)vrow * S_ + voff);

    for (int it = 0; it < S_ / 32; ++it) {
        const int kv0 = it * 32;
        const int cur = it & 1;
        if (it + 1 < S_ / 32) {
            const int nb = cur ^ 1, kvn = kv0 + 32;
            cp16(&ldsK[nb][krow * 64 + koff],
                 Kb + (size_t)(kvn + krow) * DH_ + koff);
            cp16(&ldsV[nb][vrow * 32 + voff],
                 Vb + (size_t)vrow * S_ + kvn + voff);
            ASYNC_WAIT(2);
        } else {
            ASYNC_WAIT(0);
        }
        __syncthreads();

        // scores: two 16x16 C tiles (kv groups 0-15 / 16-31), K = dh = 64
        v8f s0, s1;
        for (int g = 0; g < 8; ++g) { s0[g] = 0.0f; s1[g] = 0.0f; }
        for (int j = 0; j < 2; ++j) {
            FragBF kf0, kf1;   // B: lane = kv col, K = dh {8h..8h+7,16+8h..}
            const unsigned short* k0p = &ldsK[cur][ln * 64 + j * 32 + 8 * hh];
            const unsigned short* k1p = &ldsK[cur][(16 + ln) * 64 + j * 32 + 8 * hh];
            kf0.u[0] = *(const u32x4*)(k0p);
            kf0.u[1] = *(const u32x4*)(k0p + 16);
            kf1.u[0] = *(const u32x4*)(k1p);
            kf1.u[1] = *(const u32x4*)(k1p + 16);
            s0 = wmma_bf16(qf[j], kf0, s0);
            s1 = wmma_bf16(qf[j], kf1, s1);
        }

        // mask + online softmax.  C layout: lane = kv col ln(+16), VGPR g =
        // row M = g + 8*hh -> row stats reduce across 16 lanes.
        for (int g = 0; g < 8; ++g) {
            int qrow = q0 + g + 8 * hh;
            const int* mp = mask + ((size_t)b * S_ + qrow) * S_ + kv0;
            float v0 = s0[g] * scale; if (mp[ln] == 0)      v0 = -1e9f;
            float v1 = s1[g] * scale; if (mp[16 + ln] == 0) v1 = -1e9f;
            float cand = fmaxf(v0, v1);
            for (int o = 1; o < 16; o <<= 1)
                cand = fmaxf(cand, __shfl_xor(cand, o, 32));
            float mn    = fmaxf(mrow[g], cand);
            float alpha = __expf(mrow[g] - mn);
            float p0 = __expf(v0 - mn), p1 = __expf(v1 - mn);
            float ps = p0 + p1;
            for (int o = 1; o < 16; o <<= 1)
                ps += __shfl_xor(ps, o, 32);
            lrow[g] = lrow[g] * alpha + ps;
            mrow[g] = mn;
            for (int d = 0; d < 4; ++d) O[d][g] *= alpha;
            unsigned pp = pack2(p0, p1);   // one packed cvt for both cols
            ldsP[wave][(g + 8 * hh) * 32 + ln]      = (unsigned short)pp;
            ldsP[wave][(g + 8 * hh) * 32 + 16 + ln] = (unsigned short)(pp >> 16);
        }

        // P(16x32) @ V(32x64): A from per-wave LDS, B from V^T tile
        FragBF pf;
        const unsigned short* pb = &ldsP[wave][ln * 32];
        pf.u[0] = *(const u32x4*)(pb + 8 * hh);
        pf.u[1] = *(const u32x4*)(pb + 16 + 8 * hh);
        for (int d = 0; d < 4; ++d) {
            FragBF vf;   // lane = dh col, K = kv {8h..8h+7,16+8h..}
            const unsigned short* vb = &ldsV[cur][(d * 16 + ln) * 32 + 8 * hh];
            vf.u[0] = *(const u32x4*)(vb);
            vf.u[1] = *(const u32x4*)(vb + 16);
            O[d] = wmma_bf16(pf, vf, O[d]);
        }
        __syncthreads();
    }

    // normalize and store merged-head bf16 [b][s][h*64+dh]
    for (int d = 0; d < 4; ++d)
        for (int g = 0; g < 8; ++g) {
            int qrow = q0 + g + 8 * hh;
            float val = O[d][g] / lrow[g];
            attn[((size_t)b * S_ + qrow) * D_ + h * DH_ + d * 16 + ln] = f2bf(val);
        }
}

// ---------------------------------------------------------------------------
// Kernel 4: output projection  out[4096x1024] = attn_bf @ Wo^T + bo (fp32 out)
// Same 128x128x32 double-buffered async scheme.
// ---------------------------------------------------------------------------
__global__ __launch_bounds__(256) void out_gemm(
    const unsigned short* __restrict__ Xbf, const unsigned short* __restrict__ Wo,
    const float* __restrict__ bo, float* __restrict__ out)
{
    __shared__ unsigned short ldsX[2][128 * 32];   // 8 KB x2
    __shared__ unsigned short ldsW[2][128 * 32];   // 8 KB x2

    const int tid  = threadIdx.x;
    const int wave = tid >> 5;
    const int lane = tid & 31;
    const int ln   = lane & 15;
    const int hh   = lane >> 4;
    const int m0   = blockIdx.x * 128;
    const int n0   = blockIdx.y * 128;
    const int mq   = wave & 3;
    const int nh   = wave >> 2;

    const int xrow = tid >> 2, xo = (tid & 3) * 8;   // + i*64 rows, i=0..1

    v8f acc[2][4];
    for (int a = 0; a < 2; ++a)
        for (int i = 0; i < 4; ++i)
            for (int g = 0; g < 8; ++g) acc[a][i][g] = 0.0f;

    for (int i = 0; i < 2; ++i) {
        cp16(&ldsX[0][(xrow + i * 64) * 32 + xo],
             Xbf + (size_t)(m0 + xrow + i * 64) * D_ + xo);
        cp16(&ldsW[0][(xrow + i * 64) * 32 + xo],
             Wo + (size_t)(n0 + xrow + i * 64) * D_ + xo);
    }

    for (int it = 0; it < D_ / 32; ++it) {
        const int cur = it & 1;
        if (it + 1 < D_ / 32) {
            const int nk = (it + 1) * 32, nb = cur ^ 1;
            for (int i = 0; i < 2; ++i) {
                cp16(&ldsX[nb][(xrow + i * 64) * 32 + xo],
                     Xbf + (size_t)(m0 + xrow + i * 64) * D_ + nk + xo);
                cp16(&ldsW[nb][(xrow + i * 64) * 32 + xo],
                     Wo + (size_t)(n0 + xrow + i * 64) * D_ + nk + xo);
            }
            ASYNC_WAIT(4);
        } else {
            ASYNC_WAIT(0);
        }
        __syncthreads();

        const unsigned short* xb = ldsX[cur];
        const unsigned short* wb = ldsW[cur];
        FragBF af[2];
        for (int a = 0; a < 2; ++a) {
            const unsigned short* ab = xb + ((mq * 2 + a) * 16 + ln) * 32;
            af[a].u[0] = *(const u32x4*)(ab + 8 * hh);
            af[a].u[1] = *(const u32x4*)(ab + 16 + 8 * hh);
        }
        for (int i = 0; i < 4; ++i) {
            FragBF bf;
            const unsigned short* bb = wb + (nh * 64 + i * 16 + ln) * 32;
            bf.u[0] = *(const u32x4*)(bb + 8 * hh);
            bf.u[1] = *(const u32x4*)(bb + 16 + 8 * hh);
            acc[0][i] = wmma_bf16(af[0], bf, acc[0][i]);
            acc[1][i] = wmma_bf16(af[1], bf, acc[1][i]);
        }
        __syncthreads();
    }

    for (int a = 0; a < 2; ++a)
        for (int i = 0; i < 4; ++i) {
            int n = n0 + nh * 64 + i * 16 + ln;
            float bn = bo[n];
            for (int g = 0; g < 8; ++g) {
                int m = m0 + (mq * 2 + a) * 16 + g + 8 * hh;
                out[(size_t)m * D_ + n] = acc[a][i][g] + bn;
            }
        }
}

// ---------------------------------------------------------------------------
extern "C" void kernel_launch(void* const* d_in, const int* in_sizes, int n_in,
                              void* d_out, int out_size, void* d_ws, size_t ws_size,
                              hipStream_t stream) {
    (void)in_sizes; (void)n_in; (void)out_size; (void)ws_size;
    const float* query = (const float*)d_in[0];
    const float* key_  = (const float*)d_in[1];
    const float* value = (const float*)d_in[2];
    const int*   mask  = (const int*)d_in[3];
    const float* Wq = (const float*)d_in[4];
    const float* bq = (const float*)d_in[5];
    const float* Wk = (const float*)d_in[6];
    const float* bk = (const float*)d_in[7];
    const float* Wv = (const float*)d_in[8];
    const float* bv = (const float*)d_in[9];
    const float* Wo = (const float*)d_in[10];
    const float* bo = (const float*)d_in[11];
    float* out = (float*)d_out;

    const size_t DD  = (size_t)D_ * D_;          // 1M elems
    const size_t BSD = (size_t)B_ * S_ * D_;     // 4M elems
    unsigned short* ws   = (unsigned short*)d_ws;
    unsigned short* Wbf  = ws;                   // 4*DD bf16
    unsigned short* Xbf  = Wbf + 4 * DD;         // 3*BSD (q,k,v converted)
    unsigned short* Qp   = Xbf + 3 * BSD;        // BSD
    unsigned short* Kp   = Qp + BSD;             // BSD
    unsigned short* Vt   = Kp + BSD;             // BSD
    unsigned short* attn = Xbf;                  // alias: Xbf dead after qkv_gemm

    convert_weights<<<dim3((unsigned)(DD / 4 / 256), 4, 1), 256, 0, stream>>>(
        Wq, Wk, Wv, Wo, Wbf);
    convert_inputs<<<dim3((unsigned)(BSD / 4 / 256), 3, 1), 256, 0, stream>>>(
        query, key_, value, Xbf);

    qkv_gemm<<<dim3(B_ * S_ / 128, D_ / 128, 3), 256, 0, stream>>>(
        Xbf, Wbf, bq, bk, bv, Qp, Kp, Vt);

    attn_kernel<<<dim3(B_ * H_, S_ / 128, 1), 256, 0, stream>>>(
        Qp, Kp, Vt, mask, attn);

    out_gemm<<<dim3(B_ * S_ / 128, D_ / 128, 1), 256, 0, stream>>>(
        attn, Wbf + 3 * DD, bo, out);
}
